// bayesian_classifier_chain_clf_24404004176551
// MI455X (gfx1250) — matrix operand order repaired and checked by reading, not compiled, not run on hardware
//
#include <hip/hip_runtime.h>
#include <hip/hip_bf16.h>

typedef __attribute__((ext_vector_type(16))) __bf16 v16bf;
typedef __attribute__((ext_vector_type(8)))  float  v8f;

#define CC 64      // N_CLASSES
#define HH 256     // HIDDEN_DIM
#define G3 768     // 3*H
#define DD 512     // INPUT_DIM
#define CD 576     // C + D (W_ih row length)
#define KAUG 320   // H + C (augmented K)
#define HA_STRIDE 328
#define BS_COLS 40 // 32 used + 8 pad
#define NROWS 128  // rows (b,c pairs) per block

// ---------------- fragment helpers ----------------
__device__ __forceinline__ v16bf ldfrag_split(const __bf16* p0, const __bf16* p1) {
  union { v16bf v; uint4 q[2]; } u;
  u.q[0] = *(const uint4*)p0;
  u.q[1] = *(const uint4*)p1;
  return u.v;
}
__device__ __forceinline__ v16bf ldfrag32(const __bf16* p, int ld) {
  const int h = (threadIdx.x >> 4) & 1;
  v16bf v;
#pragma unroll
  for (int s = 0; s < 8; ++s) {
    v[s] = p[(size_t)(8 * h + s) * ld];
    v[s + 8] = p[(size_t)(16 + 8 * h + s) * ld];
  }
  return v;
}
__device__ __forceinline__ v8f wmma_bf16(v16bf a, v16bf b, v8f c) {
  return __builtin_amdgcn_wmma_f32_16x16x32_bf16(false, a, false, b, (short)0, c,
                                                 false, false);
}
__device__ __forceinline__ float sigm(float x) { return 1.0f / (1.0f + __expf(-x)); }
__device__ __forceinline__ float tanhfast(float x) {
  x = fminf(15.0f, fmaxf(-15.0f, x));
  const float e = __expf(2.0f * x);
  return (e - 1.0f) / (e + 1.0f);
}

// CDNA5 async memory->LDS copy (ASYNCcnt-tracked), GV addressing.
// dst_lds: LDS byte address (low 32 bits of flat shared pointer), 16B aligned.
__device__ __forceinline__ void async_ld_lds_b128(unsigned dst_lds, const void* src) {
  asm volatile("global_load_async_to_lds_b128 %0, %1, off"
               :: "v"(dst_lds), "v"((unsigned long long)(uintptr_t)src)
               : "memory");
}
__device__ __forceinline__ void wait_asynccnt0() {
  asm volatile("s_wait_asynccnt 0" ::: "memory");
}

// ---------------- precompute kernels ----------------
__global__ void k_cvt_bf16(const float* __restrict__ in, __bf16* __restrict__ out, int n) {
  const int i = blockIdx.x * 256 + threadIdx.x;
  if (i < n) out[i] = (__bf16)in[i];
}

// WxT[k][n] = W_ih[n][64+k]   (512 x 768, bf16, row = K)
__global__ void k_build_wxt(const float* __restrict__ Wih, __bf16* __restrict__ WxT) {
  const int i = blockIdx.x * 256 + threadIdx.x;
  if (i < DD * G3) {
    const int k = i / G3, n = i % G3;
    WxT[i] = (__bf16)Wih[(size_t)n * CD + CC + k];
  }
}

// Bfull[t][k][n]:  k<256 -> W_hh[n][k]   ;   k>=256 -> W_ih[n][P[k-256][t]]
__global__ void k_build_B(const float* __restrict__ Whh, const float* __restrict__ Wih,
                          const int* __restrict__ P, __bf16* __restrict__ B) {
  const int i = blockIdx.x * 256 + threadIdx.x;
  if (i < 3 * KAUG * G3) {
    const int t = i / (KAUG * G3);
    const int r = i % (KAUG * G3);
    const int k = r / G3, n = r % G3;
    float v;
    if (k < HH) v = Whh[(size_t)n * HH + k];
    else        v = Wih[(size_t)n * CD + P[(k - HH) * 3 + t]];
    B[i] = (__bf16)v;
  }
}

// ---------------- gx = X @ Wx^T + b_ih   (2048 x 768) ----------------
__global__ __launch_bounds__(256) void k_gx(const __bf16* __restrict__ Xb,
                                            const __bf16* __restrict__ WxT,
                                            const float* __restrict__ bih,
                                            float* __restrict__ gxout) {
  const int tid = threadIdx.x;
  const int lane = tid & 31, w = tid >> 5;
  const int half = lane >> 4, ln = lane & 15;
  const int mbase = blockIdx.x * 128 + w * 16;
  for (int chunk = 0; chunk < 6; ++chunk) {   // 6 x 128 N-cols
    v8f acc[8];
#pragma unroll
    for (int nt = 0; nt < 8; ++nt)
#pragma unroll
      for (int j = 0; j < 8; ++j) acc[nt][j] = 0.0f;
#pragma unroll 4
    for (int kt = 0; kt < 16; ++kt) {
      const __bf16* ap = Xb + (size_t)(mbase + ln) * DD + kt * 32 + half * 8;
      const v16bf a = ldfrag_split(ap, ap + 16);
#pragma unroll
      for (int nt = 0; nt < 8; ++nt) {
        const __bf16* bp = WxT + (size_t)(kt * 32) * G3 + chunk * 128 + nt * 16 + ln;
        acc[nt] = wmma_bf16(a, ldfrag32(bp, G3), acc[nt]);
      }
    }
#pragma unroll
    for (int nt = 0; nt < 8; ++nt) {
      const int n = chunk * 128 + nt * 16 + ln;
      const float bi = bih[n];
#pragma unroll
      for (int j = 0; j < 8; ++j) {
        const int m = mbase + j + half * 8;
        gxout[(size_t)m * G3 + n] = acc[nt][j] + bi;
      }
    }
  }
}

// ---------------- main 3-step recurrence + logits ----------------
__global__ __launch_bounds__(256) void k_main(
    const float* __restrict__ Y, const float* __restrict__ Wih,
    const float* __restrict__ bhh, const float* __restrict__ Wout,
    const float* __restrict__ bout, const int* __restrict__ P,
    const unsigned char* __restrict__ Mstep, const float* __restrict__ My,
    const float* __restrict__ gx, const __bf16* __restrict__ Bfull,
    float* __restrict__ out)
{
  __shared__ __align__(16) __bf16 hA[NROWS][HA_STRIDE];        // h (bf16) + one-hot cols
  __shared__ __align__(16) __bf16 Bs[2][3][KAUG][BS_COLS];     // double-buffered B chunk
  __shared__ float gx_s[2][G3];
  __shared__ float bhh_s[G3];
  __shared__ float yv_s[NROWS];
  __shared__ int   p_s[CC];
  __shared__ float ms_s[CC];

  const int tid  = threadIdx.x;
  const int lane = tid & 31;
  const int w    = tid >> 5;        // 8 waves, 16 rows each
  const int half = lane >> 4;
  const int ln   = lane & 15;
  const int m0   = blockIdx.x * NROWS;
  const int rw0  = w * 16;
  const int bl   = w >> 2;          // which of the 2 gx rows this wave uses

  // init: h = 0 (incl. one-hot columns), stage gx rows and b_hh
  for (int i = tid; i < NROWS * HA_STRIDE / 2; i += 256) ((unsigned int*)hA)[i] = 0u;
  for (int i = tid; i < 2 * G3; i += 256) {
    const int r = i / G3, col = i % G3;
    gx_s[r][col] = gx[(size_t)(blockIdx.x * 2 + r) * G3 + col];
  }
  for (int i = tid; i < G3; i += 256) bhh_s[i] = bhh[i];
  __syncthreads();

  float hreg[16][8];   // new h held in regs until step's GEMMs complete

  for (int t = 0; t < 3; ++t) {
    if (tid < CC) {
      p_s[tid]  = P[tid * 3 + t];
      ms_s[tid] = Mstep[tid * 3 + t] ? 1.0f : 0.0f;
    }
    if (tid < NROWS) {
      const int row = tid;
      const int c   = row & 63;
      const int bg  = (m0 + row) >> 6;
      const int pc  = P[c * 3 + t];
      const float yv = Y[(size_t)bg * CC + pc] * My[c * 3 + t];
      yv_s[row] = yv;
      hA[row][HH + c] = (__bf16)yv;   // one-hot augmented K column
    }
    const __bf16* Bt = Bfull + (size_t)t * KAUG * G3;

    // prologue: async-stage chunk 0 into buffer 0 (15 b128 issues/thread)
    for (int i = tid; i < 3 * KAUG * 4; i += 256) {
      const int g = i / (KAUG * 4);
      const int r = i % (KAUG * 4);
      const int k = r >> 2;
      const int q = r & 3;
      async_ld_lds_b128((unsigned)(uintptr_t)&Bs[0][g][k][q * 8],
                        Bt + (size_t)k * G3 + g * HH + q * 8);
    }
    wait_asynccnt0();
    __syncthreads();

#pragma unroll
    for (int chunk = 0; chunk < 8; ++chunk) {   // 8 x 32 cols per gate
      const int buf = chunk & 1;
      // prefetch next chunk into the other buffer; overlaps with WMMAs below
      if (chunk < 7) {
        for (int i = tid; i < 3 * KAUG * 4; i += 256) {
          const int g = i / (KAUG * 4);
          const int r = i % (KAUG * 4);
          const int k = r >> 2;
          const int q = r & 3;
          async_ld_lds_b128((unsigned)(uintptr_t)&Bs[buf ^ 1][g][k][q * 8],
                            Bt + (size_t)k * G3 + g * HH + (chunk + 1) * 32 + q * 8);
        }
      }
#pragma unroll
      for (int nt = 0; nt < 2; ++nt) {
        v8f accr, accz, accn;
#pragma unroll
        for (int j = 0; j < 8; ++j) { accr[j] = 0.f; accz[j] = 0.f; accn[j] = 0.f; }
#pragma unroll
        for (int kt = 0; kt < 10; ++kt) {       // K = 320 (r,z), 256 (n)
          const int kb = kt * 32;
          const __bf16* ap = &hA[rw0 + ln][kb + half * 8];
          const v16bf a = ldfrag_split(ap, ap + 16);
          accr = wmma_bf16(a, ldfrag32(&Bs[buf][0][kb][nt * 16 + ln], BS_COLS), accr);
          accz = wmma_bf16(a, ldfrag32(&Bs[buf][1][kb][nt * 16 + ln], BS_COLS), accz);
          if (kt < 8)
            accn = wmma_bf16(a, ldfrag32(&Bs[buf][2][kb][nt * 16 + ln], BS_COLS), accn);
        }
        const int i0 = chunk * 32 + nt * 16 + ln;     // col in [0,256)
        const float gxr = gx_s[bl][i0];
        const float gxz = gx_s[bl][HH + i0];
        const float gxn = gx_s[bl][2 * HH + i0];
        const float bhr = bhh_s[i0];
        const float bhz = bhh_s[HH + i0];
        const float bhn = bhh_s[2 * HH + i0];
#pragma unroll
        for (int j = 0; j < 8; ++j) {
          const int row  = rw0 + j + half * 8;
          const int c    = row & 63;
          const float hold = (float)hA[row][i0];
          const float r = sigm(accr[j] + gxr + bhr);
          const float z = sigm(accz[j] + gxz + bhz);
          const float wpn = Wih[(size_t)(2 * HH + i0) * CD + p_s[c]];
          const float gin = yv_s[row] * wpn + gxn;            // gi_n (input part)
          const float nn  = tanhfast(gin + r * (accn[j] + bhn));
          const float hnew = (1.0f - z) * nn + z * hold;
          hreg[chunk * 2 + nt][j] = (ms_s[c] != 0.0f) ? hnew : hold;
        }
      }
      // next-chunk data complete + all waves done reading Bs[buf]
      wait_asynccnt0();
      __syncthreads();
    }
    // write updated h back (own rows only), cols 0..255
#pragma unroll
    for (int q = 0; q < 16; ++q) {
      const int col = q * 16 + ln;
#pragma unroll
      for (int j = 0; j < 8; ++j)
        hA[rw0 + j + half * 8][col] = (__bf16)hreg[q][j];
    }
    __syncthreads();
  }

  // logits[m] = h[m] . W_out[c] + b_out[c]
  if (tid < NROWS) {
    const int row = tid;
    const int m = m0 + row;
    const int c = m & 63;
    float s = bout[c];
    const float* wo = Wout + (size_t)c * HH;
    for (int k = 0; k < HH; ++k) s += (float)hA[row][k] * wo[k];
    out[m] = s;
  }
}

// ---------------- launch ----------------
extern "C" void kernel_launch(void* const* d_in, const int* in_sizes, int n_in,
                              void* d_out, int out_size, void* d_ws, size_t ws_size,
                              hipStream_t stream) {
  (void)in_sizes; (void)n_in; (void)out_size; (void)ws_size;
  const float* X    = (const float*)d_in[0];
  const float* Y    = (const float*)d_in[1];
  const float* Wih  = (const float*)d_in[2];
  const float* Whh  = (const float*)d_in[3];
  const float* bih  = (const float*)d_in[4];
  const float* bhh  = (const float*)d_in[5];
  const float* Wout = (const float*)d_in[6];
  const float* bout = (const float*)d_in[7];
  const int*   P    = (const int*)d_in[8];
  const unsigned char* Ms = (const unsigned char*)d_in[9];
  const float* My   = (const float*)d_in[10];
  float* out = (float*)d_out;

  char* ws = (char*)d_ws;
  float*  gx    = (float*)(ws);                 // 2048*768*4      = 6,291,456
  __bf16* Xb    = (__bf16*)(ws + 6291456);      // 2048*512*2      = 2,097,152
  __bf16* WxT   = (__bf16*)(ws + 8388608);      // 512*768*2       =   786,432
  __bf16* Bfull = (__bf16*)(ws + 9175040);      // 3*320*768*2     = 1,474,560

  const int nX = 2048 * 512;
  k_cvt_bf16<<<(nX + 255) / 256, 256, 0, stream>>>(X, Xb, nX);
  k_build_wxt<<<(DD * G3 + 255) / 256, 256, 0, stream>>>(Wih, WxT);
  k_build_B<<<(3 * KAUG * G3 + 255) / 256, 256, 0, stream>>>(Whh, Wih, P, Bfull);
  k_gx<<<16, 256, 0, stream>>>(Xb, WxT, bih, gx);
  k_main<<<1024, 256, 0, stream>>>(Y, Wih, bhh, Wout, bout, P, Ms, My, gx, Bfull, out);
}
